// GATFusionBlockPosOnly_54065048322202
// MI455X (gfx1250) — matrix operand order repaired and checked by reading, not compiled, not run on hardware
//
#include <hip/hip_runtime.h>

// ---------------- constants ----------------
#define Bb 8
#define Nn 1024
#define Dd 256
#define Hg 8      // GAT heads
#define Hc 4      // cross-attn heads
#define DH 64     // Dd / Hc
#define HD 2048   // Hg * Dd

#define ALPHA 0.2f
#define NEG_GAT -9.0e15f
#define NEG_CA  -1.0e9f
#define THRESH 0.97f
#define CA_SCALE 0.125f    // 1/sqrt(64)
#define INV_SQRT_D 0.0625f // 1/sqrt(256)

// ---------------- bf16 helpers ----------------
typedef __bf16 v16bf __attribute__((ext_vector_type(16)));
typedef float  v8f   __attribute__((ext_vector_type(8)));

union FragBF {
    v16bf v;
    unsigned short us[16];
    unsigned int   u[8];
};

__device__ __forceinline__ unsigned short f2bf(float f) {
    unsigned int u = __float_as_uint(f);
    u += 0x7FFFu + ((u >> 16) & 1u);       // round-to-nearest-even
    return (unsigned short)(u >> 16);
}

// ---------------- block reductions (blockDim.x == 256) ----------------
__device__ __forceinline__ float redSum256(float v, float* sm) {
    int t = threadIdx.x;
    sm[t] = v; __syncthreads();
    for (int s = 128; s > 0; s >>= 1) { if (t < s) sm[t] += sm[t + s]; __syncthreads(); }
    float r = sm[0]; __syncthreads();
    return r;
}
__device__ __forceinline__ float redMax256(float v, float* sm) {
    int t = threadIdx.x;
    sm[t] = v; __syncthreads();
    for (int s = 128; s > 0; s >>= 1) { if (t < s) sm[t] = fmaxf(sm[t], sm[t + s]); __syncthreads(); }
    float r = sm[0]; __syncthreads();
    return r;
}

// ---------------- bf16 WMMA GEMM: C = A * Bt^T ----------------
// A [M,K] bf16 row-major (lda), Bt [Nc,K] bf16 row-major (ldb) = B transposed,
// C [M,Nc] f32 (ldc). Both operands have contiguous K per lane -> b128 loads.
// Block = 4 waves of 32; each wave computes one 32x32 C tile (4 WMMA / K-step,
// 2x fragment reuse). M, Nc multiples of 32; K multiple of 32.
__global__ __launch_bounds__(128) void k_gemm_bt(
    const unsigned short* __restrict__ A, const unsigned short* __restrict__ Bt,
    float* __restrict__ C, int M, int Nc, int K, int lda, int ldb, int ldc)
{
    const int tilesN = Nc >> 5;
    const int total  = (M >> 5) * tilesN;
    const int tile   = blockIdx.x * 4 + threadIdx.y;
    if (tile >= total) return;                    // wave-uniform
    const int tm = tile / tilesN;
    const int tn = tile - tm * tilesN;
    const int row0 = tm << 5, col0 = tn << 5;

    const int lane = threadIdx.x;
    const int r    = lane & 15;
    const int hi   = lane >> 4;
    const int kA   = hi ? 8 : 0;                  // A frag K shift for upper half-wave
    const int kB   = hi ? 16 : 0;                 // unused: B frag uses kA, same K order as A

    const unsigned short* Ar0 = A  + (size_t)(row0 + r)      * lda;
    const unsigned short* Ar1 = A  + (size_t)(row0 + 16 + r) * lda;
    const unsigned short* Br0 = Bt + (size_t)(col0 + r)      * ldb;
    const unsigned short* Br1 = Bt + (size_t)(col0 + 16 + r) * ldb;

    FragBF a0, a1, b0, b1;
    v8f c00 = {0.f,0.f,0.f,0.f,0.f,0.f,0.f,0.f};
    v8f c01 = {0.f,0.f,0.f,0.f,0.f,0.f,0.f,0.f};
    v8f c10 = {0.f,0.f,0.f,0.f,0.f,0.f,0.f,0.f};
    v8f c11 = {0.f,0.f,0.f,0.f,0.f,0.f,0.f,0.f};

    for (int k0 = 0; k0 < K; k0 += 32) {
        #pragma unroll
        for (int p = 0; p < 8; ++p) {
            // A fragment: VGPR p holds K = k0 + ((p&4)?16:0) + kA + {0..7 pairwise}
            int ka = k0 + ((p & 4) ? 16 : 0) + kA + ((p << 1) & 7);
            a0.u[p] = *(const unsigned int*)(Ar0 + ka);
            a1.u[p] = *(const unsigned int*)(Ar1 + ka);
            // B fragment (from Bt rows): same K order as A fragment
            int kb = ka;
            b0.u[p] = *(const unsigned int*)(Br0 + kb);
            b1.u[p] = *(const unsigned int*)(Br1 + kb);
        }
        c00 = __builtin_amdgcn_wmma_f32_16x16x32_bf16(false, a0.v, false, b0.v, (short)0, c00, false, false);
        c01 = __builtin_amdgcn_wmma_f32_16x16x32_bf16(false, a0.v, false, b1.v, (short)0, c01, false, false);
        c10 = __builtin_amdgcn_wmma_f32_16x16x32_bf16(false, a1.v, false, b0.v, (short)0, c10, false, false);
        c11 = __builtin_amdgcn_wmma_f32_16x16x32_bf16(false, a1.v, false, b1.v, (short)0, c11, false, false);
    }

    // C layout per 16x16 tile: VGPR i -> row (hi?8:0)+i, col = lane&15
    float* Crow = C + (size_t)(row0 + (hi ? 8 : 0)) * ldc + col0 + r;
    #pragma unroll
    for (int i = 0; i < 8; ++i) {
        Crow[(size_t)i * ldc]              = c00[i];
        Crow[(size_t)i * ldc + 16]         = c01[i];
        Crow[(size_t)(i + 16) * ldc]       = c10[i];
        Crow[(size_t)(i + 16) * ldc + 16]  = c11[i];
    }
}

// ---------------- elementwise / reduction kernels ----------------

__global__ void k_f2bf(const float* __restrict__ src, unsigned short* __restrict__ dst, int n) {
    int i = blockIdx.x * 256 + threadIdx.x;
    if (i < n) dst[i] = f2bf(src[i]);
}

// batched f32->bf16 transpose: src [batch][rows][cols] -> dst [batch][cols][rows]
__global__ void k_f2bfT(const float* __restrict__ src, unsigned short* __restrict__ dst,
                        int rows, int cols, int total) {
    int idx = blockIdx.x * 256 + threadIdx.x;
    if (idx >= total) return;
    int rc = rows * cols;
    int bi = idx / rc, rem = idx - bi * rc;
    int r = rem / cols, c = rem - r * cols;
    dst[(size_t)bi * rc + (size_t)c * rows + r] = f2bf(src[idx]);
}

// pos_query[b,:] = masked mean of x over tokens with mask==1
__global__ void k_posquery(const float* __restrict__ x, const int* __restrict__ mask,
                           float* __restrict__ pq) {
    int b = blockIdx.x, e = threadIdx.x;
    float s = 0.f; int cnt = 0;
    for (int n = 0; n < Nn; ++n) {
        int m = (mask[b * Nn + n] == 1);
        cnt += m;
        if (m) s += x[((size_t)b * Nn + n) * Dd + e];
    }
    pq[b * Dd + e] = s / fmaxf((float)cnt, 1.0f);
}

// t = sim_Wx @ (pos_query @ sim_Wq): u[e] = sum_d pq[d]*Wq[d,e]; t[d] = sum_e Wx[d,e]*u[e]
__global__ void k_tvec(const float* __restrict__ pq, const float* __restrict__ Wx,
                       const float* __restrict__ Wq, float* __restrict__ tvec) {
    __shared__ float u[Dd];
    int b = blockIdx.x, t = threadIdx.x;
    float s = 0.f;
    for (int d0 = 0; d0 < Dd; ++d0) s += pq[b * Dd + d0] * Wq[d0 * Dd + t];
    u[t] = s; __syncthreads();
    float s2 = 0.f;
    for (int e = 0; e < Dd; ++e) s2 += Wx[t * Dd + e] * u[e];
    tvec[b * Dd + t] = s2;
}

// pos_sim = sigmoid(x . t / sqrt(d)); sel = pos_sim > THRESH
__global__ void k_possim(const float* __restrict__ x, const float* __restrict__ tvec,
                         float* __restrict__ simout, int* __restrict__ sel) {
    int bn = blockIdx.x * 256 + threadIdx.x;   // [0, B*N)
    int b = bn >> 10;
    const float* xr = x + (size_t)bn * Dd;
    const float* tv = tvec + b * Dd;
    float s = 0.f;
    for (int e = 0; e < Dd; ++e) s += xr[e] * tv[e];
    float sim = 1.f / (1.f + expf(-s * INV_SQRT_D));
    simout[bn] = sim;
    sel[bn] = (sim > THRESH) ? 1 : 0;
}

// Wh post: store TRANSPOSED bf16 copy (for Bt GEMM); f1/f2 row dots.  grid(N,Hg)
__global__ void k_whpost(const float* __restrict__ Whf, const float* __restrict__ a1,
                         const float* __restrict__ a2, unsigned short* __restrict__ WhbT,
                         float* __restrict__ f1, float* __restrict__ f2) {
    __shared__ float sm[256];
    int n = blockIdx.x, h = blockIdx.y, t = threadIdx.x;
    float v = Whf[((size_t)h * Nn + n) * Dd + t];
    WhbT[((size_t)h * Dd + t) * Nn + n] = f2bf(v);   // [h][e][n]
    float s1 = redSum256(v * a1[h * Dd + t], sm);
    float s2 = redSum256(v * a2[h * Dd + t], sm);
    if (t == 0) { f1[h * Nn + n] = s1; f2[h * Nn + n] = s2; }
}

// GAT masked softmax over rank-1 logits; connect = sel[n]&sel[m]&(gram>0).  grid(N,Hg)
__global__ void k_gat_softmax(const float* __restrict__ gram, const int* __restrict__ selb,
                              const float* __restrict__ f1, const float* __restrict__ f2,
                              unsigned short* __restrict__ probs) {
    __shared__ float sm[256];
    int n = blockIdx.x, h = blockIdx.y, t = threadIdx.x;
    int seln = selb[n];
    float fr = f1[h * Nn + n];
    float vals[4];
    float mx = -3.4e38f;
    #pragma unroll
    for (int j = 0; j < 4; ++j) {
        int m = t + j * 256;
        float z = fr + f2[h * Nn + m];
        z = (z >= 0.f) ? z : ALPHA * z;
        bool c = seln && selb[m] && (gram[(size_t)n * Nn + m] > 0.f);
        vals[j] = c ? z : NEG_GAT;
        mx = fmaxf(mx, vals[j]);
    }
    mx = redMax256(mx, sm);
    float ls = 0.f;
    #pragma unroll
    for (int j = 0; j < 4; ++j) { vals[j] = expf(vals[j] - mx); ls += vals[j]; }
    float rinv = 1.f / redSum256(ls, sm);
    unsigned short* pr = probs + ((size_t)h * Nn + n) * Nn;
    #pragma unroll
    for (int j = 0; j < 4; ++j) pr[t + j * 256] = f2bf(vals[j] * rinv);
}

// ELU + concat heads: cat[n, h*Dd+e] = elu(hbuf[h,n,e])   grid(Hg*Nn)
__global__ void k_hpost(const float* __restrict__ hbuf, unsigned short* __restrict__ cat) {
    int idx = blockIdx.x * 256 + threadIdx.x;        // Hg*Nn*Dd = 2^21
    int h = idx >> 18;
    int rem = idx & ((1 << 18) - 1);
    int n = rem >> 8, e = rem & 255;
    float v = hbuf[idx];
    v = (v > 0.f) ? v : (expf(v) - 1.f);
    cat[(size_t)n * HD + h * Dd + e] = f2bf(v);
}

// Who post: TRANSPOSED bf16 copy + eo1/eo2 row dots    grid(N)
__global__ void k_whopost(const float* __restrict__ Whof, const float* __restrict__ ao1,
                          const float* __restrict__ ao2, unsigned short* __restrict__ WhobT,
                          float* __restrict__ eo1, float* __restrict__ eo2) {
    __shared__ float sm[256];
    int n = blockIdx.x, t = threadIdx.x;
    float v = Whof[(size_t)n * Dd + t];
    WhobT[(size_t)t * Nn + n] = f2bf(v);             // [e][n]
    float s1 = redSum256(v * ao1[t], sm);
    float s2 = redSum256(v * ao2[t], sm);
    if (t == 0) { eo1[n] = s1; eo2[n] = s2; }
}

// GAT output-layer masked softmax (single head)    grid(N)
__global__ void k_o_softmax(const float* __restrict__ gram, const int* __restrict__ selb,
                            const float* __restrict__ eo1, const float* __restrict__ eo2,
                            unsigned short* __restrict__ probs_o) {
    __shared__ float sm[256];
    int n = blockIdx.x, t = threadIdx.x;
    int seln = selb[n];
    float fr = eo1[n];
    float vals[4];
    float mx = -3.4e38f;
    #pragma unroll
    for (int j = 0; j < 4; ++j) {
        int m = t + j * 256;
        float z = fr + eo2[m];
        z = (z >= 0.f) ? z : ALPHA * z;
        bool c = seln && selb[m] && (gram[(size_t)n * Nn + m] > 0.f);
        vals[j] = c ? z : NEG_GAT;
        mx = fmaxf(mx, vals[j]);
    }
    mx = redMax256(mx, sm);
    float ls = 0.f;
    #pragma unroll
    for (int j = 0; j < 4; ++j) { vals[j] = expf(vals[j] - mx); ls += vals[j]; }
    float rinv = 1.f / redSum256(ls, sm);
    unsigned short* pr = probs_o + (size_t)n * Nn;
    #pragma unroll
    for (int j = 0; j < 4; ++j) pr[t + j * 256] = f2bf(vals[j] * rinv);
}

// kv = LN4(sel ? gout + pos_emb : 0) -> bf16      grid(N)
__global__ void k_kvln(const float* __restrict__ gout, const float* __restrict__ pe,
                       const int* __restrict__ selb, const float* __restrict__ g,
                       const float* __restrict__ bt, unsigned short* __restrict__ out) {
    __shared__ float sm[256];
    int n = blockIdx.x, t = threadIdx.x;
    float v = selb[n] ? (gout[(size_t)n * Dd + t] + pe[(size_t)n * Dd + t]) : 0.f;
    float mean = redSum256(v, sm) * (1.f / Dd);
    float d0 = v - mean;
    float var = redSum256(d0 * d0, sm) * (1.f / Dd);
    float y = d0 * rsqrtf(var + 1e-5f) * g[t] + bt[t];
    out[(size_t)n * Dd + t] = f2bf(y);
}

// qx = LN3(x_row) -> bf16      grid(N)
__global__ void k_lnrow(const float* __restrict__ in, const float* __restrict__ g,
                        const float* __restrict__ bt, unsigned short* __restrict__ out) {
    __shared__ float sm[256];
    int n = blockIdx.x, t = threadIdx.x;
    float v = in[(size_t)n * Dd + t];
    float mean = redSum256(v, sm) * (1.f / Dd);
    float d0 = v - mean;
    float var = redSum256(d0 * d0, sm) * (1.f / Dd);
    float y = d0 * rsqrtf(var + 1e-5f) * g[t] + bt[t];
    out[(size_t)n * Dd + t] = f2bf(y);
}

// [N,Dd] f32 -> per-head [Hc][N][DH] bf16 (A-operand layout)     grid(N)
__global__ void k_split(const float* __restrict__ src, unsigned short* __restrict__ dst) {
    int idx = blockIdx.x * 256 + threadIdx.x;   // N*Dd
    int n = idx >> 8, col = idx & 255;
    int h = col >> 6, c = col & 63;
    dst[((size_t)h * Nn + n) * DH + c] = f2bf(src[idx]);
}

// [N,Dd] f32 -> per-head TRANSPOSED [Hc][DH][N] bf16 (Bt-operand layout)   grid(N)
__global__ void k_splitT(const float* __restrict__ src, unsigned short* __restrict__ dst) {
    int idx = blockIdx.x * 256 + threadIdx.x;
    int n = idx >> 8, col = idx & 255;
    int h = col >> 6, c = col & 63;
    dst[((size_t)h * DH + c) * Nn + n] = f2bf(src[idx]);
}

// CA masked softmax    grid(N, Hc)
__global__ void k_ca_softmax(const float* __restrict__ s_log, const int* __restrict__ selb,
                             unsigned short* __restrict__ pca) {
    __shared__ float sm[256];
    int n = blockIdx.x, h = blockIdx.y, t = threadIdx.x;
    const float* srow = s_log + ((size_t)h * Nn + n) * Nn;
    float vals[4];
    float mx = -3.4e38f;
    #pragma unroll
    for (int j = 0; j < 4; ++j) {
        int m = t + j * 256;
        vals[j] = selb[m] ? srow[m] * CA_SCALE : NEG_CA;
        mx = fmaxf(mx, vals[j]);
    }
    mx = redMax256(mx, sm);
    float ls = 0.f;
    #pragma unroll
    for (int j = 0; j < 4; ++j) { vals[j] = expf(vals[j] - mx); ls += vals[j]; }
    float rinv = 1.f / redSum256(ls, sm);
    unsigned short* pr = pca + ((size_t)h * Nn + n) * Nn;
    #pragma unroll
    for (int j = 0; j < 4; ++j) pr[t + j * 256] = f2bf(vals[j] * rinv);
}

// [Hc][N][DH] f32 -> [N,Dd] bf16     grid(N)
__global__ void k_merge(const float* __restrict__ obuf, unsigned short* __restrict__ ocat) {
    int idx = blockIdx.x * 256 + threadIdx.x;
    int n = idx >> 8, col = idx & 255;
    int h = col >> 6, c = col & 63;
    ocat[idx] = f2bf(obuf[((size_t)h * Nn + n) * DH + c]);
}

// out = x + gamma * oproj     grid(N)
__global__ void k_final(const float* __restrict__ xf, const float* __restrict__ oproj,
                        const float* __restrict__ gamma, float* __restrict__ outb) {
    int idx = blockIdx.x * 256 + threadIdx.x;
    outb[idx] = xf[idx] + gamma[idx & 255] * oproj[idx];
}

// ---------------- host launcher ----------------
static inline void gemm(const unsigned short* A, const unsigned short* Bt, float* C,
                        int M, int Nc, int K, int lda, int ldb, int ldc, hipStream_t s) {
    int tiles = (M >> 5) * (Nc >> 5);
    dim3 blk(32, 4);
    int grid = (tiles + 3) / 4;
    k_gemm_bt<<<grid, blk, 0, s>>>(A, Bt, C, M, Nc, K, lda, ldb, ldc);
}

extern "C" void kernel_launch(void* const* d_in, const int* in_sizes, int n_in,
                              void* d_out, int out_size, void* d_ws, size_t ws_size,
                              hipStream_t stream) {
    (void)in_sizes; (void)n_in; (void)out_size; (void)ws_size;

    const float* x       = (const float*)d_in[0];
    const int*   mask    = (const int*)  d_in[1];
    const float* pos_emb = (const float*)d_in[2];
    const float* sim_Wx  = (const float*)d_in[3];
    const float* sim_Wq  = (const float*)d_in[4];
    const float* adj_W   = (const float*)d_in[5];
    const float* gat_W   = (const float*)d_in[6];
    const float* gat_a1  = (const float*)d_in[7];
    const float* gat_a2  = (const float*)d_in[8];
    const float* gat_Wo  = (const float*)d_in[9];
    const float* gat_ao1 = (const float*)d_in[10];
    const float* gat_ao2 = (const float*)d_in[11];
    const float* ln3_g   = (const float*)d_in[12];
    const float* ln3_b   = (const float*)d_in[13];
    const float* ln4_g   = (const float*)d_in[14];
    const float* ln4_b   = (const float*)d_in[15];
    const float* ca_Wq   = (const float*)d_in[16];
    const float* ca_Wk   = (const float*)d_in[17];
    const float* ca_Wv   = (const float*)d_in[18];
    const float* ca_Wp   = (const float*)d_in[19];
    const float* gamma   = (const float*)d_in[20];

    float* out_x   = (float*)d_out;
    float* out_sim = out_x + (size_t)Bb * Nn * Dd;

    // -------- workspace carve --------
    char* base = (char*)d_ws;
    size_t off = 0;
    auto alloc = [&](size_t bytes) -> void* {
        void* p = base + off;
        off += (bytes + 255) & ~(size_t)255;
        return p;
    };
    unsigned short* xb_all  = (unsigned short*)alloc((size_t)Bb * Nn * Dd * 2);
    unsigned short* adjWbT  = (unsigned short*)alloc((size_t)Dd * Dd * 2);
    unsigned short* gatWbT  = (unsigned short*)alloc((size_t)Hg * Dd * Dd * 2);
    unsigned short* gatWobT = (unsigned short*)alloc((size_t)HD * Dd * 2);
    unsigned short* WqbT    = (unsigned short*)alloc((size_t)Dd * Dd * 2);
    unsigned short* WkbT    = (unsigned short*)alloc((size_t)Dd * Dd * 2);
    unsigned short* WvbT    = (unsigned short*)alloc((size_t)Dd * Dd * 2);
    unsigned short* WpbT    = (unsigned short*)alloc((size_t)Dd * Dd * 2);
    float* pq   = (float*)alloc((size_t)Bb * Dd * 4);
    float* tvec = (float*)alloc((size_t)Bb * Dd * 4);
    int*   sel  = (int*)  alloc((size_t)Bb * Nn * 4);
    // per-sample scratch (reused across the b loop)
    float*          fa_f32  = (float*)alloc((size_t)Nn * Dd * 4);          // aliased: obuf (Hc*Nn*DH f32)
    unsigned short* fa_bf   = (unsigned short*)alloc((size_t)Nn * Dd * 2);
    float*          gram    = (float*)alloc((size_t)Nn * Nn * 4);
    float*          Whf     = (float*)alloc((size_t)Hg * Nn * Dd * 4);     // aliased: pca (Hc*Nn*Nn bf16)
    unsigned short* WhbT    = (unsigned short*)alloc((size_t)Hg * Nn * Dd * 2); // [h][e][n]
    float*          f1      = (float*)alloc((size_t)Hg * Nn * 4);
    float*          f2      = (float*)alloc((size_t)Hg * Nn * 4);
    unsigned short* probs   = (unsigned short*)alloc((size_t)Hg * Nn * Nn * 2); // aliased: s_log (Hc*Nn*Nn f32)
    float*          hbuf    = (float*)alloc((size_t)Hg * Nn * Dd * 4);
    unsigned short* catb    = (unsigned short*)alloc((size_t)Nn * HD * 2);
    float*          Whof    = (float*)alloc((size_t)Nn * Dd * 4);
    unsigned short* WhobT   = (unsigned short*)alloc((size_t)Nn * Dd * 2);      // [e][n]
    float*          eo1     = (float*)alloc((size_t)Nn * 4);
    float*          eo2     = (float*)alloc((size_t)Nn * 4);
    unsigned short* probs_o = (unsigned short*)alloc((size_t)Nn * Nn * 2);
    float*          gout    = (float*)alloc((size_t)Nn * Dd * 4);
    unsigned short* qxb     = (unsigned short*)alloc((size_t)Nn * Dd * 2);
    unsigned short* kvb     = (unsigned short*)alloc((size_t)Nn * Dd * 2);
    float*          projf   = (float*)alloc((size_t)Nn * Dd * 4);          // reused for q/k/v proj and oproj
    unsigned short* qb      = (unsigned short*)alloc((size_t)Hc * Nn * DH * 2); // [h][n][c]
    unsigned short* kb      = (unsigned short*)alloc((size_t)Hc * Nn * DH * 2); // [h][n][c] (Bt for logits)
    unsigned short* vbT     = (unsigned short*)alloc((size_t)Hc * Nn * DH * 2); // [h][c][n] (Bt for A@V)
    unsigned short* ocat    = (unsigned short*)alloc((size_t)Nn * Dd * 2);
    // aliases of dead regions (lifetimes verified per-sample)
    float*          s_log = (float*)(void*)probs;        // CA logits, after GAT probs consumed
    unsigned short* pca   = (unsigned short*)(void*)Whf; // CA probs, after Whf consumed
    float*          obuf  = fa_f32;                      // CA head outputs, after fa consumed
    float*          oproj = projf;

    // -------- one-time (per call) conversions + selection pipeline --------
    k_f2bf<<<(Bb * Nn * Dd) / 256, 256, 0, stream>>>(x, xb_all, Bb * Nn * Dd);
    k_f2bfT<<<(Dd * Dd) / 256, 256, 0, stream>>>(adj_W, adjWbT, Dd, Dd, Dd * Dd);
    k_f2bfT<<<(Hg * Dd * Dd) / 256, 256, 0, stream>>>(gat_W, gatWbT, Dd, Dd, Hg * Dd * Dd);
    k_f2bfT<<<(HD * Dd) / 256, 256, 0, stream>>>(gat_Wo, gatWobT, HD, Dd, HD * Dd);
    k_f2bfT<<<(Dd * Dd) / 256, 256, 0, stream>>>(ca_Wq, WqbT, Dd, Dd, Dd * Dd);
    k_f2bfT<<<(Dd * Dd) / 256, 256, 0, stream>>>(ca_Wk, WkbT, Dd, Dd, Dd * Dd);
    k_f2bfT<<<(Dd * Dd) / 256, 256, 0, stream>>>(ca_Wv, WvbT, Dd, Dd, Dd * Dd);
    k_f2bfT<<<(Dd * Dd) / 256, 256, 0, stream>>>(ca_Wp, WpbT, Dd, Dd, Dd * Dd);

    k_posquery<<<Bb, 256, 0, stream>>>(x, mask, pq);
    k_tvec<<<Bb, 256, 0, stream>>>(pq, sim_Wx, sim_Wq, tvec);
    k_possim<<<(Bb * Nn) / 256, 256, 0, stream>>>(x, tvec, out_sim, sel);

    // -------- per-sample pipeline --------
    for (int b = 0; b < Bb; ++b) {
        const float*          xf   = x      + (size_t)b * Nn * Dd;
        const unsigned short* xb   = xb_all + (size_t)b * Nn * Dd;
        const int*            selb = sel    + (size_t)b * Nn;

        // adjacency: gram = (x@adj_W)(x@adj_W)^T ; sign(gram) == (adj>0.5)
        gemm(xb, adjWbT, fa_f32, Nn, Dd, Dd, Dd, Dd, Dd, stream);
        k_f2bf<<<(Nn * Dd) / 256, 256, 0, stream>>>(fa_f32, fa_bf, Nn * Dd);
        gemm(fa_bf, fa_bf, gram, Nn, Nn, Dd, Dd, Dd, Nn, stream);

        // GAT heads: Wh_h = x @ gat_W[h]
        for (int h = 0; h < Hg; ++h)
            gemm(xb, gatWbT + (size_t)h * Dd * Dd, Whf + (size_t)h * Nn * Dd,
                 Nn, Dd, Dd, Dd, Dd, Dd, stream);
        k_whpost<<<dim3(Nn, Hg), 256, 0, stream>>>(Whf, gat_a1, gat_a2, WhbT, f1, f2);
        k_gat_softmax<<<dim3(Nn, Hg), 256, 0, stream>>>(gram, selb, f1, f2, probs);
        for (int h = 0; h < Hg; ++h)
            gemm(probs + (size_t)h * Nn * Nn, WhbT + (size_t)h * Dd * Nn,
                 hbuf + (size_t)h * Nn * Dd, Nn, Dd, Nn, Nn, Nn, Dd, stream);
        k_hpost<<<Hg * Nn, 256, 0, stream>>>(hbuf, catb);

        // GAT out layer
        gemm(catb, gatWobT, Whof, Nn, Dd, HD, HD, HD, Dd, stream);
        k_whopost<<<Nn, 256, 0, stream>>>(Whof, gat_ao1, gat_ao2, WhobT, eo1, eo2);
        k_o_softmax<<<Nn, 256, 0, stream>>>(gram, selb, eo1, eo2, probs_o);
        gemm(probs_o, WhobT, gout, Nn, Dd, Nn, Nn, Nn, Dd, stream);

        // cross-attention inputs
        k_kvln<<<Nn, 256, 0, stream>>>(gout, pos_emb, selb, ln4_g, ln4_b, kvb);
        k_lnrow<<<Nn, 256, 0, stream>>>(xf, ln3_g, ln3_b, qxb);
        gemm(qxb, WqbT, projf, Nn, Dd, Dd, Dd, Dd, Dd, stream);
        k_split<<<Nn, 256, 0, stream>>>(projf, qb);
        gemm(kvb, WkbT, projf, Nn, Dd, Dd, Dd, Dd, Dd, stream);
        k_split<<<Nn, 256, 0, stream>>>(projf, kb);
        gemm(kvb, WvbT, projf, Nn, Dd, Dd, Dd, Dd, Dd, stream);
        k_splitT<<<Nn, 256, 0, stream>>>(projf, vbT);

        // attention: logits (q@k^T), softmax, A@V
        for (int h = 0; h < Hc; ++h)
            gemm(qb + (size_t)h * Nn * DH, kb + (size_t)h * Nn * DH,
                 s_log + (size_t)h * Nn * Nn, Nn, Nn, DH, DH, DH, Nn, stream);
        k_ca_softmax<<<dim3(Nn, Hc), 256, 0, stream>>>(s_log, selb, pca);
        for (int h = 0; h < Hc; ++h)
            gemm(pca + (size_t)h * Nn * Nn, vbT + (size_t)h * DH * Nn,
                 obuf + (size_t)h * Nn * DH, Nn, DH, Nn, Nn, Nn, DH, stream);
        k_merge<<<Nn, 256, 0, stream>>>(obuf, ocat);

        // output projection + residual
        gemm(ocat, WpbT, oproj, Nn, Dd, Dd, Dd, Dd, Dd, stream);
        k_final<<<Nn, 256, 0, stream>>>(xf, oproj, gamma, out_x + (size_t)b * Nn * Dd);
    }
}
